// GraphSelfAttention_58703613002244
// MI455X (gfx1250) — compile-verified, matrix operand order checked by reading, not run on hardware
//
#include <hip/hip_runtime.h>
#include <cstdint>

// ---------------------------------------------------------------------------
// GraphSelfAttention on MI455X (gfx1250).
//  * All four matmuls on V_WMMA_F32_16X16X32_BF16 (f32 accumulate).
//  * Flash-style fused attention: 268MB sim matrix never touches HBM.
//  * Weights prepacked once into fragment-native bf16 blobs (b128 loads).
//  * K/V tiles double-buffered into LDS with GLOBAL_LOAD_ASYNC_TO_LDS_B128
//    (ASYNCcnt + s_wait_asynccnt), staged cooperatively by 4 waves per block
//    so each K/V stream is reused by 64 query rows (4x less L2 traffic).
// ---------------------------------------------------------------------------

typedef __attribute__((ext_vector_type(16))) __bf16 v16bf;
typedef __attribute__((ext_vector_type(8)))  __bf16 v8bf;
typedef __attribute__((ext_vector_type(8)))  float  v8f;

#define QDIM   512
#define INNER  256
#define HEADS  4
#define DHEAD  64
#define BATCH  4
#define SEQ    2048

// ---- CDNA5 async copy + wait (inline asm; ISA 10.x / 15.18) ---------------
__device__ __forceinline__ void async_b128(unsigned lds_off, const void* gaddr) {
    asm volatile("global_load_async_to_lds_b128 %0, %1, off"
                 :: "v"(lds_off), "v"(gaddr) : "memory");
}
__device__ __forceinline__ void wait_async0() {
    asm volatile("s_wait_asynccnt 0x0" ::: "memory");
}

// ---- fragment loaders -----------------------------------------------------
// 16-bit A/B fragment: lane holds 16 K-elements; lanes 0-15 take
// K = {c..c+7, c+16..c+23} with c=0, lanes 16-31 same with c=8.
__device__ __forceinline__ v16bf frag_from_row_bf16(const __bf16* row, int kbase, int lane) {
    const int c = kbase + ((lane & 16) ? 8 : 0);
    v8bf g0 = *reinterpret_cast<const v8bf*>(row + c);
    v8bf g1 = *reinterpret_cast<const v8bf*>(row + c + 16);
    v16bf f;
#pragma unroll
    for (int i = 0; i < 8; ++i) { f[i] = g0[i]; f[i + 8] = g1[i]; }
    return f;
}

__device__ __forceinline__ v16bf frag_from_row_f32(const float* row, int kbase, int lane) {
    const int c = kbase + ((lane & 16) ? 8 : 0);
    v16bf f;
#pragma unroll
    for (int i = 0; i < 8; ++i) {
        f[i]     = (__bf16)row[c + i];
        f[i + 8] = (__bf16)row[c + 16 + i];
    }
    return f;
}

#define WMMA_BF16(A, B, C) \
    __builtin_amdgcn_wmma_f32_16x16x32_bf16(false, (A), false, (B), (short)0, (C), false, false)

// ---------------------------------------------------------------------------
// Kernel 0: prepack a row-major [Ktot x Nout] f32 weight into fragment-native
// bf16 blobs: blob[(kt*(Nout/16)+nt)*32 + lane][16].  One wave per tile.
// ---------------------------------------------------------------------------
__global__ __launch_bounds__(32) void pack_w_kernel(
    const float* __restrict__ W, __bf16* __restrict__ P, int Ktot, int Nout) {
    const int lane = threadIdx.x & 31;
    const int ntiles = Nout >> 4;
    const int tile = blockIdx.x;           // kt * ntiles + nt
    const int nt = tile % ntiles;
    const int kt = tile / ntiles;
    const int col = nt * 16 + (lane & 15);
    const int c   = kt * 32 + ((lane & 16) ? 8 : 0);
    v16bf f;
#pragma unroll
    for (int i = 0; i < 8; ++i) {
        f[i]     = (__bf16)W[(size_t)(c + i) * Nout + col];
        f[i + 8] = (__bf16)W[(size_t)(c + 16 + i) * Nout + col];
    }
    *reinterpret_cast<v16bf*>(P + ((size_t)tile * 32 + lane) * 16) = f;
}

// ---------------------------------------------------------------------------
// Kernel 1: QKV projection.  x[8192,512] @ W[512,256] -> bf16.
// grid = (512 M-tiles, 4 N-groups of 64, 3 matrices); wave = 16x64 tile,
// A-fragment reused across 4 WMMAs per k-step.  Q scaled by 1/8; V stored
// transposed [b,h,64,n].
// ---------------------------------------------------------------------------
__global__ __launch_bounds__(32) void qkv_proj_kernel(
    const float* __restrict__ x,
    const __bf16* __restrict__ Wqp, const __bf16* __restrict__ Wkp,
    const __bf16* __restrict__ Wvp,
    __bf16* __restrict__ Qh, __bf16* __restrict__ Kh, __bf16* __restrict__ Vt) {

    const int lane = threadIdx.x & 31;
    const int mt = blockIdx.x;
    const int ng = blockIdx.y;           // group of 4 column tiles
    const int w  = blockIdx.z;           // 0=Q 1=K 2=V
    const __bf16* Wp = (w == 0) ? Wqp : (w == 1) ? Wkp : Wvp;

    const float* xrow = x + (size_t)(mt * 16 + (lane & 15)) * QDIM;
    const int NT = INNER / 16;           // 16 column tiles total

    v8f acc[4] = {};
    for (int kt = 0; kt < QDIM / 32; ++kt) {
        v16bf a = frag_from_row_f32(xrow, kt * 32, lane);
#pragma unroll
        for (int j = 0; j < 4; ++j) {
            const int nt = ng * 4 + j;
            v16bf b = *reinterpret_cast<const v16bf*>(
                Wp + ((size_t)(kt * NT + nt) * 32 + lane) * 16);
            acc[j] = WMMA_BF16(a, b, acc[j]);
        }
    }

    const float qscale = (w == 0) ? 0.125f : 1.0f;   // DIM_HEAD^-0.5
    const int hf   = (lane >> 4) & 1;
    const int ncol = lane & 15;
#pragma unroll
    for (int j = 0; j < 4; ++j) {
        const int hcol = (ng * 4 + j) * 16 + ncol;
        const int hd = hcol >> 6;
        const int dd = hcol & 63;
#pragma unroll
        for (int r = 0; r < 8; ++r) {
            const int g  = mt * 16 + r + 8 * hf;
            const int bb = g >> 11;
            const int nn = g & (SEQ - 1);
            const float val = acc[j][r] * qscale;
            if (w == 2) {
                Vt[((size_t)(bb * HEADS + hd) * DHEAD + dd) * SEQ + nn] = (__bf16)val;
            } else {
                __bf16* dst = (w == 0) ? Qh : Kh;
                dst[((size_t)(bb * HEADS + hd) * SEQ + nn) * DHEAD + dd] = (__bf16)val;
            }
        }
    }
}

// ---------------------------------------------------------------------------
// Kernel 2: flash attention core.  Block = 4 waves = 64 query rows; the four
// waves share one double-buffered async-staged K/V block (32 keys) in LDS.
// grid = (32 q-supertiles, 4 heads, 4 batches).
// ---------------------------------------------------------------------------
__global__ __launch_bounds__(128) void attn_kernel(
    const __bf16* __restrict__ Qh, const __bf16* __restrict__ Kh,
    const __bf16* __restrict__ Vt, const unsigned char* __restrict__ mask,
    __bf16* __restrict__ Oi) {

    // K tile: 32 keys x 64 d (4KB, contiguous in Kh).  V tile: 64 d x 32 keys.
    __shared__ __attribute__((aligned(16))) __bf16 kbuf[2][32 * DHEAD];
    __shared__ __attribute__((aligned(16))) __bf16 vbuf[2][DHEAD * 32];
    __shared__ __attribute__((aligned(16))) __bf16 pLds[4][16 * 32];

    const int tid  = threadIdx.x;        // 0..127
    const int lane = tid & 31;
    const int wid  = tid >> 5;           // wave id: which 16-query tile
    const int hd = blockIdx.y;
    const int bb = blockIdx.z;
    const int bh = bb * HEADS + hd;
    const int q0 = (blockIdx.x * 4 + wid) * 16;   // first query row of this wave

    const __bf16* Qb = Qh + (size_t)bh * SEQ * DHEAD;
    const __bf16* Kb = Kh + (size_t)bh * SEQ * DHEAD;
    const __bf16* Vb = Vt + (size_t)bh * DHEAD * SEQ;

    // Q A-fragments (16 rows x 64 d -> 2 k-steps)
    const __bf16* qrow = Qb + (size_t)(q0 + (lane & 15)) * DHEAD;
    const v16bf qa0 = frag_from_row_bf16(qrow, 0, lane);
    const v16bf qa1 = frag_from_row_bf16(qrow, 32, lane);

    v8f acc[4] = {};
    float mrun[8], lrun[8];
#pragma unroll
    for (int r = 0; r < 8; ++r) { mrun[r] = -1.0e30f; lrun[r] = 0.0f; }

    const int hf   = (lane >> 4) & 1;
    const int ncol = lane & 15;
    const unsigned char* mrow0 = mask + ((size_t)bb * SEQ + q0) * SEQ;

    // ---- cooperative async stage of one (K,V) key-block ------------------
    // 8KB total = 512 x 16B segments; 128 threads x 4 segments each.
    auto stage = [&](int buf, int k0) {
        const char* gk = (const char*)(Kb + (size_t)k0 * DHEAD);   // 4KB block
        const unsigned lk = (unsigned)(uintptr_t)&kbuf[buf][0];
#pragma unroll
        for (int i = 0; i < 2; ++i) {
            const unsigned o = (unsigned)((i * 128 + tid) * 16);
            async_b128(lk + o, gk + o);
        }
        const unsigned lv = (unsigned)(uintptr_t)&vbuf[buf][0];
#pragma unroll
        for (int i = 0; i < 2; ++i) {
            const int seg = i * 128 + tid;         // 256 x 16B segments
            const int d   = seg >> 2;
            const int off = (seg & 3) * 8;         // elements within a 32-key row
            async_b128(lv + (unsigned)(d * 64 + off * 2),
                       (const void*)(Vb + (size_t)d * SEQ + k0 + off));
        }
    };

    const int NBLK = SEQ / 32;
    stage(0, 0);
    wait_async0();
    __syncthreads();
    int cur = 0;

    for (int jt = 0; jt < NBLK; ++jt) {
        const int k0 = jt * 32;
        if (jt + 1 < NBLK) stage(cur ^ 1, k0 + 32);   // prefetch next block

        // ---- S = Q*K^T, two 16-key tiles (K read from LDS) ---------------
        v8f s[2];
#pragma unroll
        for (int t = 0; t < 2; ++t) {
            const __bf16* krow = &kbuf[cur][(t * 16 + ncol) * DHEAD];
            v16bf kb0 = frag_from_row_bf16(krow, 0, lane);
            v16bf kb1 = frag_from_row_bf16(krow, 32, lane);
            v8f sa = {};
            sa = WMMA_BF16(qa0, kb0, sa);
            sa = WMMA_BF16(qa1, kb1, sa);
            s[t] = sa;
        }

        // ---- mask + per-row tile max -------------------------------------
        float tmax[8];
#pragma unroll
        for (int r = 0; r < 8; ++r) {
            const int q = r + 8 * hf;
            const unsigned char* mr = mrow0 + (size_t)q * SEQ + k0 + ncol;
            const float s0 = mr[0]  ? s[0][r] : -3.0e38f;
            const float s1 = mr[16] ? s[1][r] : -3.0e38f;
            s[0][r] = s0; s[1][r] = s1;
            tmax[r] = fmaxf(s0, s1);
        }
#pragma unroll
        for (int off = 1; off < 16; off <<= 1)
#pragma unroll
            for (int r = 0; r < 8; ++r)
                tmax[r] = fmaxf(tmax[r], __shfl_xor(tmax[r], off, 32));

        // ---- online softmax update ---------------------------------------
        float psum[8];
#pragma unroll
        for (int r = 0; r < 8; ++r) {
            const float mnew = fmaxf(fmaxf(mrun[r], tmax[r]), -1.0e30f);
            const float corr = __expf(mrun[r] - mnew);
            const float p0 = __expf(s[0][r] - mnew);
            const float p1 = __expf(s[1][r] - mnew);
            s[0][r] = p0; s[1][r] = p1;
            psum[r] = p0 + p1;
            mrun[r] = mnew;
            lrun[r] *= corr;
#pragma unroll
            for (int dt = 0; dt < 4; ++dt) acc[dt][r] *= corr;
        }
#pragma unroll
        for (int off = 1; off < 16; off <<= 1)
#pragma unroll
            for (int r = 0; r < 8; ++r)
                psum[r] += __shfl_xor(psum[r], off, 32);
#pragma unroll
        for (int r = 0; r < 8; ++r) lrun[r] += psum[r];

        // ---- P: C-fragment -> A-fragment via per-wave LDS tile -----------
#pragma unroll
        for (int r = 0; r < 8; ++r) {
            const int q = r + 8 * hf;
            pLds[wid][q * 32 + ncol]      = (__bf16)s[0][r];
            pLds[wid][q * 32 + 16 + ncol] = (__bf16)s[1][r];
        }
        __syncthreads();
        const v16bf pa = frag_from_row_bf16(&pLds[wid][(lane & 15) * 32], 0, lane);

        // ---- O += P * V (V read from LDS) --------------------------------
#pragma unroll
        for (int dt = 0; dt < 4; ++dt) {
            const __bf16* vrow = &vbuf[cur][(dt * 16 + ncol) * 32];
            v16bf vb = frag_from_row_bf16(vrow, 0, lane);
            acc[dt] = WMMA_BF16(pa, vb, acc[dt]);
        }

        // publish next staged buffer / retire current one, block-wide
        if (jt + 1 < NBLK) wait_async0();
        __syncthreads();
        if (jt + 1 < NBLK) cur ^= 1;
    }

    // ---- normalize + store ----------------------------------------------
#pragma unroll
    for (int r = 0; r < 8; ++r) {
        const int n  = q0 + r + 8 * hf;
        const float inv = 1.0f / lrun[r];
#pragma unroll
        for (int dt = 0; dt < 4; ++dt) {
            Oi[((size_t)bb * SEQ + n) * INNER + hd * DHEAD + dt * 16 + ncol] =
                (__bf16)(acc[dt][r] * inv);
        }
    }
}

// ---------------------------------------------------------------------------
// Kernel 3: output projection.  Oi[8192,256] @ Wo[256,512] + bo -> f32.
// wave = 16x64 tile with packed Wo fragments.
// ---------------------------------------------------------------------------
__global__ __launch_bounds__(32) void out_proj_kernel(
    const __bf16* __restrict__ Oi, const __bf16* __restrict__ Wop,
    const float* __restrict__ bo, float* __restrict__ out) {

    const int lane = threadIdx.x & 31;
    const int mt = blockIdx.x;           // 512 tiles of 16 rows
    const int ng = blockIdx.y;           // 8 groups of 64 cols
    const int NT = QDIM / 16;            // 32 column tiles

    const __bf16* arow = Oi + (size_t)(mt * 16 + (lane & 15)) * INNER;

    v8f acc[4] = {};
#pragma unroll
    for (int kt = 0; kt < INNER / 32; ++kt) {
        v16bf a = frag_from_row_bf16(arow, kt * 32, lane);
#pragma unroll
        for (int j = 0; j < 4; ++j) {
            const int nt = ng * 4 + j;
            v16bf b = *reinterpret_cast<const v16bf*>(
                Wop + ((size_t)(kt * NT + nt) * 32 + lane) * 16);
            acc[j] = WMMA_BF16(a, b, acc[j]);
        }
    }

    const int hf   = (lane >> 4) & 1;
    const int ncol = lane & 15;
#pragma unroll
    for (int j = 0; j < 4; ++j) {
        const int col = (ng * 4 + j) * 16 + ncol;
        const float bias = bo[col];
#pragma unroll
        for (int r = 0; r < 8; ++r) {
            const int g = mt * 16 + r + 8 * hf;
            out[(size_t)g * QDIM + col] = acc[j][r] + bias;
        }
    }
}

// ---------------------------------------------------------------------------
extern "C" void kernel_launch(void* const* d_in, const int* in_sizes, int n_in,
                              void* d_out, int out_size, void* d_ws, size_t ws_size,
                              hipStream_t stream) {
    const float*         x    = (const float*)d_in[0];
    const unsigned char* mask = (const unsigned char*)d_in[1];   // jnp bool -> 1 byte
    const float*         Wq   = (const float*)d_in[2];
    const float*         Wk   = (const float*)d_in[3];
    const float*         Wv   = (const float*)d_in[4];
    const float*         Wo   = (const float*)d_in[5];
    const float*         bo   = (const float*)d_in[6];

    // ws layout (bf16): Q | K | V^T | O_inner | packed Wq,Wk,Wv,Wo
    const size_t qkv_elems  = (size_t)BATCH * HEADS * SEQ * DHEAD;  // 2,097,152
    const size_t pack_elems = (size_t)QDIM * INNER;                 // 131,072
    __bf16* Qh  = (__bf16*)d_ws;
    __bf16* Kh  = Qh + qkv_elems;
    __bf16* Vt  = Kh + qkv_elems;
    __bf16* Oi  = Vt + qkv_elems;
    __bf16* Wqp = Oi + qkv_elems;
    __bf16* Wkp = Wqp + pack_elems;
    __bf16* Wvp = Wkp + pack_elems;
    __bf16* Wop = Wvp + pack_elems;

    // prepack weights into fragment-native layout (trivial cost, once per call)
    pack_w_kernel<<<dim3((QDIM / 32) * (INNER / 16)), 32, 0, stream>>>(Wq, Wqp, QDIM, INNER);
    pack_w_kernel<<<dim3((QDIM / 32) * (INNER / 16)), 32, 0, stream>>>(Wk, Wkp, QDIM, INNER);
    pack_w_kernel<<<dim3((QDIM / 32) * (INNER / 16)), 32, 0, stream>>>(Wv, Wvp, QDIM, INNER);
    pack_w_kernel<<<dim3((INNER / 32) * (QDIM / 16)), 32, 0, stream>>>(Wo, Wop, INNER, QDIM);

    dim3 g1((BATCH * SEQ) / 16, INNER / 64, 3);
    qkv_proj_kernel<<<g1, 32, 0, stream>>>(x, Wqp, Wkp, Wvp, Qh, Kh, Vt);

    dim3 g2(SEQ / 64, HEADS, BATCH);
    attn_kernel<<<g2, 128, 0, stream>>>(Qh, Kh, Vt, mask, Oi);

    dim3 g3((BATCH * SEQ) / 16, QDIM / 64);
    out_proj_kernel<<<g3, 32, 0, stream>>>(Oi, Wop, bo, (float*)d_out);
}
